// Quantizer_20736102105457
// MI455X (gfx1250) — matrix operand order repaired and checked by reading, not compiled, not run on hardware
//
#include <hip/hip_runtime.h>

// ---------------------------------------------------------------------------
// VQ nearest-neighbour quantizer for MI455X (gfx1250, wave32, WMMA).
//   z: [16, 256, 32, 32] f32  -> z_flat [N=16384, D=256]
//   codebook: [K=16384, D=256] f32
//   out[n,:] = codebook[argmin_k(||e_k||^2 - 2 z_n.e_k), :]  (f32, [N,256] flat)
// Matmul path: V_WMMA_F32_16X16X32_BF16, f32 accumulate.
// Codebook tiles staged in LDS (async global->LDS, double buffered),
// shared by all 8 waves of the workgroup; 32 z-rows resident per wave.
// ---------------------------------------------------------------------------

typedef __attribute__((ext_vector_type(16))) __bf16 v16bf;
typedef __attribute__((ext_vector_type(8)))  __bf16 v8bf;
typedef __attribute__((ext_vector_type(8)))  float  v8f;
typedef __attribute__((ext_vector_type(4)))  int    v4i;

#define N_ROWS   16384   // B*H*W = 16*32*32
#define N_CODES  16384
#define DIM      256
#define KSPLIT   2                    // gridDim.y: codebook halves
#define CODES_PER_SPLIT (N_CODES / KSPLIT)
#define TILES    (CODES_PER_SPLIT / 16)
#define LDS_PITCH 272                 // 16-elem pad: 544B = 136 dwords (==8 mod 64 banks)

#ifndef __has_builtin
#define __has_builtin(x) 0
#endif

#if __has_builtin(__builtin_amdgcn_global_load_async_to_lds_b128)
#define VQ_ASYNC 1
#else
#define VQ_ASYNC 0
#endif

// 16B global -> LDS (async copy engine; ASYNCcnt-tracked) or load+ds_store.
__device__ __forceinline__ void copy8_to_lds(const __bf16* g, __bf16* l) {
#if VQ_ASYNC
  __builtin_amdgcn_global_load_async_to_lds_b128((v4i*)g, (v4i*)l, 0, 0);
#else
  *(v8bf*)l = *(const v8bf*)g;   // load + ds_store fallback
#endif
}

__device__ __forceinline__ void wait_async_done() {
#if VQ_ASYNC
#if __has_builtin(__builtin_amdgcn_s_wait_asynccnt)
  __builtin_amdgcn_s_wait_asynccnt(0);
#else
  asm volatile("s_wait_asynccnt 0x0" ::: "memory");
#endif
#endif
}

// ---------------------------------------------------------------------------
// Kernel 1: NCHW -> [N, D] row-major bf16 transpose (LDS tiled, coalesced).
// grid (8, 32, 16) = (C/32, HW/32, B), block (32, 8)
// ---------------------------------------------------------------------------
__global__ __launch_bounds__(256) void vq_transpose_kernel(
    const float* __restrict__ z, __bf16* __restrict__ zb) {
  __shared__ float tile[32][33];
  const int b   = blockIdx.z;
  const int c0  = blockIdx.x * 32;
  const int hw0 = blockIdx.y * 32;
  const float* zp = z + (size_t)b * DIM * 1024;
#pragma unroll
  for (int i = 0; i < 4; ++i) {
    int c = c0 + threadIdx.y + i * 8;
    tile[threadIdx.y + i * 8][threadIdx.x] = zp[(size_t)c * 1024 + hw0 + threadIdx.x];
  }
  __syncthreads();
#pragma unroll
  for (int i = 0; i < 4; ++i) {
    int hw = hw0 + threadIdx.y + i * 8;
    zb[((size_t)b * 1024 + hw) * DIM + c0 + threadIdx.x] =
        (__bf16)tile[threadIdx.x][threadIdx.y + i * 8];
  }
}

// ---------------------------------------------------------------------------
// Kernel 2: codebook -> bf16 copy + per-row squared norms (f32).
// grid = K, block = 256
// ---------------------------------------------------------------------------
__global__ __launch_bounds__(256) void vq_cbprep_kernel(
    const float* __restrict__ cb, __bf16* __restrict__ cbb,
    float* __restrict__ cbn) {
  const int k = blockIdx.x;
  const int c = threadIdx.x;
  float v = cb[(size_t)k * DIM + c];
  cbb[(size_t)k * DIM + c] = (__bf16)v;
  float s = v * v;
#pragma unroll
  for (int m = 16; m >= 1; m >>= 1) s += __shfl_xor(s, m, 32);
  __shared__ float wsum[8];
  if ((threadIdx.x & 31) == 0) wsum[threadIdx.x >> 5] = s;
  __syncthreads();
  if (threadIdx.x == 0) {
    float t = 0.f;
#pragma unroll
    for (int i = 0; i < 8; ++i) t += wsum[i];
    cbn[k] = t;
  }
}

// ---------------------------------------------------------------------------
// Kernel 3: WMMA distance argmin.
//   WG = 256 threads (8 waves); wave owns 32 rows (2 A-tile sets in regs).
//   grid = (N/256, KSPLIT); each WG sweeps CODES_PER_SPLIT codes.
//   16-code tile (16 codes x 256 dims bf16) staged in LDS, double buffered,
//   loaded with async global->LDS copies, consumed by all 8 waves.
//
// Operand layouts used here (16-bit, wave32; A/B use contiguous k, not the ISA table order):
//   A (16x32): lane m (0..15): row m, K = 0..15 contiguous; lane m+16: K = 16..31.
//   B (32x16): lane l (0..15): col l, K = 0..15 contiguous; lane l+16: K = 16..31.
//   C/D (16x16 f32): VGPR r: lanes 0..15 -> M=r, lanes 16..31 -> M=r+8.
// ---------------------------------------------------------------------------
__global__ __launch_bounds__(256) void vq_argmin_kernel(
    const __bf16* __restrict__ zb, const __bf16* __restrict__ cbb,
    const float* __restrict__ cbn, float* __restrict__ bv_part,
    int* __restrict__ bi_part) {
  __shared__ __bf16 bbuf[2][16 * LDS_PITCH];   // 2 x 8704 B

  const int tid  = threadIdx.x;
  const int lane = tid & 31;
  const int wave = tid >> 5;
  const int half = lane >> 4;        // K-half selector
  const int l15  = lane & 15;
  const int m0   = blockIdx.x * 256 + wave * 32;
  const int cbase = blockIdx.y * CODES_PER_SPLIT;
  const int yoff  = blockIdx.y * N_ROWS;

  // staging assignment: thread -> (code row r, 16-elem segment s)
  const int sr = tid >> 4;           // 0..15
  const int ss = tid & 15;           // 0..15
  const __bf16* gstage = cbb + (size_t)(cbase + sr) * DIM + ss * 16;
  __bf16* lstage0 = &bbuf[0][sr * LDS_PITCH + ss * 16];
  __bf16* lstage1 = &bbuf[1][sr * LDS_PITCH + ss * 16];

  // ---- load A: 2 sets x 8 chunks of 16x32 bf16, resident all sweep ----
  v16bf a0[8], a1[8];
  const __bf16* ab0 = zb + (size_t)(m0 + l15) * DIM + half * 16;
  const __bf16* ab1 = zb + (size_t)(m0 + 16 + l15) * DIM + half * 16;
#pragma unroll
  for (int kc = 0; kc < 8; ++kc) {
    v8bf lo0 = *(const v8bf*)(ab0 + kc * 32);
    v8bf hi0 = *(const v8bf*)(ab0 + kc * 32 + 8);
    v8bf lo1 = *(const v8bf*)(ab1 + kc * 32);
    v8bf hi1 = *(const v8bf*)(ab1 + kc * 32 + 8);
#pragma unroll
    for (int i = 0; i < 8; ++i) {
      a0[kc][i] = lo0[i]; a0[kc][i + 8] = hi0[i];
      a1[kc][i] = lo1[i]; a1[kc][i + 8] = hi1[i];
    }
  }

  float best0[8], best1[8];
  int   bidx0[8], bidx1[8];
#pragma unroll
  for (int r = 0; r < 8; ++r) {
    best0[r] = 3.4e38f; bidx0[r] = 0;
    best1[r] = 3.4e38f; bidx1[r] = 0;
  }

  // prologue: stage tile 0 into buffer 0
  copy8_to_lds(gstage, lstage0);
  copy8_to_lds(gstage + 8, lstage0 + 8);
  wait_async_done();
  __syncthreads();

  for (int t = 0; t < TILES; ++t) {
    const int c0 = cbase + t * 16;

    // prefetch next tile into the other buffer (overlaps WMMA below)
    if (t + 1 < TILES) {
      const __bf16* g = gstage + (size_t)(t + 1) * 16 * DIM;
      __bf16* l = ((t + 1) & 1) ? lstage1 : lstage0;
      copy8_to_lds(g, l);
      copy8_to_lds(g + 8, l + 8);
    }

    // ---- 16x16 dot tile vs 32 rows: 16 WMMAs on LDS-resident B ----
    const __bf16* bb = &bbuf[t & 1][l15 * LDS_PITCH + half * 16];
    v8f acc0 = {0.f, 0.f, 0.f, 0.f, 0.f, 0.f, 0.f, 0.f};
    v8f acc1 = {0.f, 0.f, 0.f, 0.f, 0.f, 0.f, 0.f, 0.f};
#pragma unroll
    for (int kc = 0; kc < 8; ++kc) {
      v16bf bmat = *(const v16bf*)(bb + kc * 32);
      acc0 = __builtin_amdgcn_wmma_f32_16x16x32_bf16(
          false, a0[kc], false, bmat, (short)0, acc0, false, false);
      acc1 = __builtin_amdgcn_wmma_f32_16x16x32_bf16(
          false, a1[kc], false, bmat, (short)0, acc1, false, false);
    }

    const int   code = c0 + l15;
    const float cn   = cbn[code];
#pragma unroll
    for (int r = 0; r < 8; ++r) {
      float s0 = cn - 2.0f * acc0[r];      // ||e||^2 - 2 z.e (||z||^2 const)
      if (s0 < best0[r]) { best0[r] = s0; bidx0[r] = code; }
      float s1 = cn - 2.0f * acc1[r];
      if (s1 < best1[r]) { best1[r] = s1; bidx1[r] = code; }
    }

    if (t + 1 < TILES) {
      wait_async_done();   // next tile landed
      __syncthreads();     // all waves done reading buf[t&1]; all parts visible
    }
  }

  // ---- cross-lane min/argmin over the 16 lanes sharing each row ----
#pragma unroll
  for (int r = 0; r < 8; ++r) {
    float v  = best0[r];
    int   id = bidx0[r];
#pragma unroll
    for (int m = 8; m >= 1; m >>= 1) {
      float ov = __shfl_xor(v, m, 32);
      int   oi = __shfl_xor(id, m, 32);
      if (ov < v || (ov == v && oi < id)) { v = ov; id = oi; }
    }
    if (l15 == 0) {
      int row = m0 + half * 8 + r;
      bv_part[yoff + row] = v;
      bi_part[yoff + row] = id;
    }

    v  = best1[r];
    id = bidx1[r];
#pragma unroll
    for (int m = 8; m >= 1; m >>= 1) {
      float ov = __shfl_xor(v, m, 32);
      int   oi = __shfl_xor(id, m, 32);
      if (ov < v || (ov == v && oi < id)) { v = ov; id = oi; }
    }
    if (l15 == 0) {
      int row = m0 + 16 + half * 8 + r;
      bv_part[yoff + row] = v;
      bi_part[yoff + row] = id;
    }
  }
}

// ---------------------------------------------------------------------------
// Kernel 4: merge split-K partials + gather f32 codebook row.
// grid = N, block = 256
// ---------------------------------------------------------------------------
__global__ __launch_bounds__(256) void vq_merge_gather_kernel(
    const float* __restrict__ cb, const float* __restrict__ bv,
    const int* __restrict__ bi, float* __restrict__ out) {
  const int n = blockIdx.x;
  float v0 = bv[n], v1 = bv[N_ROWS + n];
  int   i0 = bi[n], i1 = bi[N_ROWS + n];
  int id = (v1 < v0 || (v1 == v0 && i1 < i0)) ? i1 : i0;
  out[(size_t)n * DIM + threadIdx.x] = cb[(size_t)id * DIM + threadIdx.x];
}

// ---------------------------------------------------------------------------
extern "C" void kernel_launch(void* const* d_in, const int* in_sizes, int n_in,
                              void* d_out, int out_size, void* d_ws, size_t ws_size,
                              hipStream_t stream) {
  const float* z  = (const float*)d_in[0];   // [16, 256, 32, 32]
  const float* cb = (const float*)d_in[1];   // [16384, 256]
  float* out = (float*)d_out;                // [16384, 256] flat

  char* ws = (char*)d_ws;
  __bf16* zb  = (__bf16*)ws;                                       // 8 MB
  __bf16* cbb = (__bf16*)(ws + (size_t)8 * 1024 * 1024);           // 8 MB
  float*  cbn = (float*) (ws + (size_t)16 * 1024 * 1024);          // 64 KB
  float*  bvp = (float*) (ws + (size_t)16 * 1024 * 1024 + (64 << 10));   // 128 KB
  int*    bip = (int*)   (ws + (size_t)16 * 1024 * 1024 + (192 << 10));  // 128 KB

  vq_transpose_kernel<<<dim3(DIM / 32, 1024 / 32, 16), dim3(32, 8), 0, stream>>>(z, zb);
  vq_cbprep_kernel<<<N_CODES, 256, 0, stream>>>(cb, cbb, cbn);
  vq_argmin_kernel<<<dim3(N_ROWS / 256, KSPLIT), 256, 0, stream>>>(zb, cbb, cbn, bvp, bip);
  vq_merge_gather_kernel<<<N_ROWS, 256, 0, stream>>>(cb, bvp, bip, out);
}
